// HeadsOp_2525440770314
// MI455X (gfx1250) — matrix operand order repaired and checked by reading, not compiled, not run on hardware
//
#include <hip/hip_runtime.h>

// CDNA5 / gfx1250 windowed-attention kernel (Swin-style), wave32 + WMMA.
// One 128-thread (4-wave) workgroup per (window b, head h).

typedef __attribute__((ext_vector_type(16))) _Float16 v16h;
typedef __attribute__((ext_vector_type(8)))  _Float16 v8h;
typedef __attribute__((ext_vector_type(4)))  _Float16 v4h;
typedef __attribute__((ext_vector_type(8)))  float    v8f;

#define HEADS 24
#define HDIM  32
#define NTOK  49
#define NPAD  64
#define NWIN  64
#define SUPER 1024
#define CH3   3072          // 3 * SUPER
#define LOG2E 1.4426950408889634f
// Q pre-scale: (1/sqrt(32)) * log2(e), so softmax exponent is a bare exp2.
#define QK_SCALE (0.17677669529663687f * 1.4426950408889634f)

#define SHUF16(lo, hi) __builtin_shufflevector(lo, hi, 0,1,2,3,4,5,6,7,8,9,10,11,12,13,14,15)

// ds_swizzle group-of-32 mode: offset = {xor[14:10], or[9:5], and[4:0]}
#define SWZ_XOR(x, xmask) \
    __int_as_float(__builtin_amdgcn_ds_swizzle(__float_as_int(x), (0x1f | ((xmask) << 10))))

#if __has_builtin(__builtin_amdgcn_exp2f)
#define EXP2F(x) __builtin_amdgcn_exp2f(x)
#else
#define EXP2F(x) __expf((x) * 0.6931471805599453f)
#endif

// Precomputed padded (bias + mask) * log2e for all 64x24 (window, head) pairs.
// 25 MB zero-init device global: recomputed from the inputs on every launch,
// L2-resident (192 MB L2) during the main kernel.
__device__ float g_bm[(size_t)NWIN * HEADS * NPAD * NPAD];

// ---------------------------------------------------------------------------
// bm[w][h][n][m], n,m in [0,64): -inf for m>=49 (n<49), 0 for n>=49.
// ---------------------------------------------------------------------------
__launch_bounds__(128)
__global__ void bias_precompute_kernel(const float* __restrict__ mask,
                                       const float* __restrict__ rpb,
                                       const int*   __restrict__ rpi)
{
    const int wh = blockIdx.x;                 // w*HEADS + h
    const int w  = wh / HEADS;
    const int h  = wh - w * HEADS;
    const float* maskw = mask + (size_t)w * (NTOK * NTOK);
    float* dst = g_bm + ((size_t)wh << 12);    // 4096 floats per (w,h)
    for (int i = threadIdx.x; i < NPAD * NPAD; i += 128) {
        const int n = i >> 6;
        const int m = i & 63;
        float v;
        if (n >= NTOK)      v = 0.0f;
        else if (m >= NTOK) v = -__builtin_inff();
        else {
            const int nm = n * NTOK + m;
            v = (rpb[rpi[nm] * 32 + h] + maskw[nm]) * LOG2E;
        }
        dst[i] = v;
    }
}

__launch_bounds__(128)
__global__ void swin_attn_kernel(const float* __restrict__ qkv,
                                 float* __restrict__ out)
{
    __shared__ __align__(16) _Float16 sQ [NPAD * HDIM];   // [n][d] f16, pre-scaled, rows>=49 zero
    __shared__ __align__(16) _Float16 sK [NPAD * HDIM];   // [m][d] f16, rows>=49 zero
    __shared__ __align__(16) _Float16 sVt[HDIM * NPAD];   // [d][m] f16, m 49..63 zeroed
    __shared__ __align__(16) _Float16 sP [NPAD * NPAD];   // softmax probs f16
    __shared__ __align__(16) float    sB [NPAD * NPAD];   // (bias+mask)*log2e, -inf / 0 padded

    const int bh  = blockIdx.x;
    const int b   = bh / HEADS;
    const int h   = bh - b * HEADS;
    const int tid = threadIdx.x;

    // ---------------- stage Q, K, V(-transposed) into LDS as f16 ----------------
    const float* base = qkv + (size_t)b * (NTOK * CH3);
    for (int i = tid; i < NTOK * 8; i += 128) {
        const int row = i >> 3, seg = i & 7;
        const float* rp = base + (size_t)row * CH3 + h * HDIM + seg * 4;
        const float4 fq = *(const float4*)(rp);
        const float4 fk = *(const float4*)(rp + 768);
        const float4 fv = *(const float4*)(rp + 1536);
        v4h hq = { (_Float16)(fq.x * QK_SCALE), (_Float16)(fq.y * QK_SCALE),
                   (_Float16)(fq.z * QK_SCALE), (_Float16)(fq.w * QK_SCALE) };
        v4h hk = { (_Float16)fk.x, (_Float16)fk.y, (_Float16)fk.z, (_Float16)fk.w };
        *(v4h*)&sQ[row * HDIM + seg * 4] = hq;
        *(v4h*)&sK[row * HDIM + seg * 4] = hk;
        const int d0 = seg * 4;
        sVt[(d0 + 0) * NPAD + row] = (_Float16)fv.x;
        sVt[(d0 + 1) * NPAD + row] = (_Float16)fv.y;
        sVt[(d0 + 2) * NPAD + row] = (_Float16)fv.z;
        sVt[(d0 + 3) * NPAD + row] = (_Float16)fv.w;
    }
    // zero Q/K padding rows 49..63 (contiguous 480 f16 = 60 x 16B)
    if (tid < 60) {
        v8h z8 = {};
        *(v8h*)&sQ[NTOK * HDIM + tid * 8] = z8;
        *(v8h*)&sK[NTOK * HDIM + tid * 8] = z8;
    }
    // zero V-transpose padding columns m = 49..63 (these feed real outputs)
    for (int i = tid; i < HDIM * (NPAD - NTOK); i += 128) {
        const int d = i / (NPAD - NTOK);
        const int m = NTOK + (i - d * (NPAD - NTOK));
        sVt[d * NPAD + m] = (_Float16)0.0f;
    }
    // ---- stage combined (bias + mask) * log2e tile: straight L2 -> LDS copy ----
    {
        const float4* src = (const float4*)(g_bm + (((size_t)(b & 63) * HEADS + h) << 12));
        for (int i = tid; i < 1024; i += 128)
            *(float4*)&sB[i * 4] = src[i];
    }
    __syncthreads();

    const int wave = tid >> 5;
    const int lane = tid & 31;
    const int half = lane >> 4;      // which 16-lane half
    const int l16  = lane & 15;

    // ---------------- S = Q * K^T   (this wave: rows 16*wave .. 16*wave+15) -----
    // A fragment (16x32 f16): lane<16 holds K{0..7,16..23}, lane>=16 K{8..15,24..31}
    const int arow = 16 * wave + l16;
    const v8h qlo = *(const v8h*)&sQ[arow * HDIM + half * 8];
    const v8h qhi = *(const v8h*)&sQ[arow * HDIM + 16 + half * 8];
    const v16h aQ = SHUF16(qlo, qhi);

    v8f c[4];
    #pragma unroll
    for (int ct = 0; ct < 4; ++ct) {
        // B fragment (32x16 f16): col = l16 of tile, lane<16 holds K{0..7,16..23}, lane>=16 K{8..15,24..31}
        const int m = 16 * ct + l16;
        const v8h klo = *(const v8h*)&sK[m * HDIM + half * 8];
        const v8h khi = *(const v8h*)&sK[m * HDIM + 16 + half * 8];
        const v16h bK = SHUF16(klo, khi);
        v8f z = {};
        c[ct] = __builtin_amdgcn_wmma_f32_16x16x32_f16(
                    false, aQ, false, bK, (short)0, z, false, false);
    }

    // ---------------- softmax (no max-shift: |logit| bounded ~25 in exp2 domain)
    // C/D layout: vgpr r -> row (r + 8*half) of tile strip, lane%16 -> column
    #pragma unroll
    for (int r = 0; r < 8; ++r) {
        const int n = 16 * wave + 8 * half + r;
        const float* sBr = &sB[n << 6];
        float sum = 0.0f;
        #pragma unroll
        for (int ct = 0; ct < 4; ++ct) {
            const float p = EXP2F(c[ct][r] + sBr[16 * ct + l16]); // -inf -> 0 on pads
            c[ct][r] = p;
            sum += p;
        }
        // row sum across the 16-lane half: ds_swizzle xor reduction
        sum += SWZ_XOR(sum, 1);
        sum += SWZ_XOR(sum, 2);
        sum += SWZ_XOR(sum, 4);
        sum += SWZ_XOR(sum, 8);
        const float inv = 1.0f / sum;
        #pragma unroll
        for (int ct = 0; ct < 4; ++ct)
            sP[(n << 6) + 16 * ct + l16] = (_Float16)(c[ct][r] * inv);
    }
    __syncthreads();

    // ---------------- O = P * V  (K-dim = 64 padded, 2 chunks of 32) -----------
    v8f o[2] = {};
    const int prow = 16 * wave + l16;
    #pragma unroll
    for (int chunk = 0; chunk < 2; ++chunk) {
        const int kb = chunk * 32;
        const v8h plo = *(const v8h*)&sP[prow * NPAD + kb + half * 8];
        const v8h phi = *(const v8h*)&sP[prow * NPAD + kb + 16 + half * 8];
        const v16h aP = SHUF16(plo, phi);
        #pragma unroll
        for (int dt = 0; dt < 2; ++dt) {
            const int d = 16 * dt + l16;
            const v8h vlo = *(const v8h*)&sVt[d * NPAD + kb + half * 8];
            const v8h vhi = *(const v8h*)&sVt[d * NPAD + kb + 16 + half * 8];
            const v16h bV = SHUF16(vlo, vhi);
            o[dt] = __builtin_amdgcn_wmma_f32_16x16x32_f16(
                        false, aP, false, bV, (short)0, o[dt], false, false);
        }
    }

    // ---------------- store O into padded embedding ----------------------------
    float* outb = out + (size_t)b * (NTOK * SUPER) + h * HDIM;
    #pragma unroll
    for (int r = 0; r < 8; ++r) {
        const int n = 16 * wave + 8 * half + r;
        if (n < NTOK) {
            outb[(size_t)n * SUPER + 0  + l16] = o[0][r];
            outb[(size_t)n * SUPER + 16 + l16] = o[1][r];
        }
    }
}

// Zero the supernet padding channels [768, 1024) for every (b, n).
__global__ void zero_pad_kernel(float* __restrict__ out)
{
    const int total = 1024 * NTOK * 64;           // float4 count: B*N*256/4
    for (int j = blockIdx.x * blockDim.x + threadIdx.x; j < total;
         j += gridDim.x * blockDim.x) {
        const int bn = j >> 6;
        const int c4 = j & 63;
        float4 z = make_float4(0.f, 0.f, 0.f, 0.f);
        *(float4*)&out[(size_t)bn * SUPER + 768 + c4 * 4] = z;
    }
}

extern "C" void kernel_launch(void* const* d_in, const int* in_sizes, int n_in,
                              void* d_out, int out_size, void* d_ws, size_t ws_size,
                              hipStream_t stream)
{
    (void)in_sizes; (void)n_in; (void)out_size; (void)d_ws; (void)ws_size;
    const float* qkv  = (const float*)d_in[0];   // [1024, 49, 3072] f32
    const float* mask = (const float*)d_in[1];   // [64, 49, 49] f32
    const float* rpb  = (const float*)d_in[2];   // [169, 32] f32
    const int*   rpi  = (const int*)  d_in[3];   // [2401] i32
    float* out = (float*)d_out;                  // [1024, 49, 1024] f32

    bias_precompute_kernel<<<NWIN * HEADS, 128, 0, stream>>>(mask, rpb, rpi);
    zero_pad_kernel<<<2048, 256, 0, stream>>>(out);
    swin_attn_kernel<<<1024 * HEADS, 128, 0, stream>>>(qkv, out);
}
